// BidGRU_3977139716150
// MI455X (gfx1250) — compile-verified, matrix operand order checked by reading, not run on hardware
//
#include <hip/hip_runtime.h>

// ---------------------------------------------------------------------------
// BiGRU + FC for MI455X (gfx1250), wave32, WMMA bf16 16x16x32 + TDM staging.
//
// Shapes: VOCAB=32000, EMB=128, HID=256, B=8, T=256.
//
// Pipeline (all on `stream`, all state in d_ws):
//   1) pack_bf16_tiles : fp32 W -> bf16 tiles in WMMA B-matrix lane layout
//   2) embed_gather    : x_bf[t*8+b][e] = bf16(embedding[inputs[b][t]][e])
//   3) proj_kernel     : xg[z][t][b][n] = x @ Wx[z] + bias[z]   (WMMA,
//                        B tile staged to LDS once per block via TDM)
//   4) scan_kernel     : 2 blocks (fwd/bwd), 16 waves, LDS-resident h,
//                        per step: r,z gates (WMMA) -> barrier ->
//                        h~=tanh((r*h)@Whh+xh) -> h update -> hcat bf16,
//                        prefetch of next step's xg slices
//   5) fc_kernel       : out = hcat @ fc_w^T + fc_b              (WMMA,
//                        16KB B tile staged to LDS via TDM, shared by 8 waves)
//
// WMMA operand layouts (CDNA5 ISA 7.12.2):
//   A 16x32 bf16: lane L holds row M=L%16; element e -> k = kc*32 +
//                 (e/8)*16 + ((L&16)?8:0) + (e%8)  => two 16B loads per lane.
//   B 32x16 bf16: pre-packed so lane L's 16 elements are contiguous 32B.
//   C/D 16x16 f32: lane L col N=L%16, VGPR v row M = v + 8*(L/16).
// ---------------------------------------------------------------------------

typedef __attribute__((ext_vector_type(16))) __bf16 v16bf;
typedef __attribute__((ext_vector_type(8)))  __bf16 v8bf;
typedef __attribute__((ext_vector_type(8)))  float  v8f;
typedef __attribute__((ext_vector_type(4)))  unsigned int u32x4;
typedef __attribute__((ext_vector_type(8)))  int i32x8;
typedef __attribute__((ext_vector_type(4)))  int i32x4;

#if defined(__has_builtin)
#if __has_builtin(__builtin_amdgcn_tensor_load_to_lds)
#define HAVE_TDM 1
#endif
#endif

static __device__ __forceinline__ v8f wmma_bf16(v16bf a, v16bf b, v8f c) {
  return __builtin_amdgcn_wmma_f32_16x16x32_bf16(false, a, false, b, (short)0, c,
                                                 false, false);
}

// A-matrix fragment load from row-major bf16 buffer (global or LDS; rs must be
// a multiple of 8 elements for 16B alignment).
static __device__ __forceinline__ v16bf load_a_frag(const __bf16* base, int rs,
                                                    int row, int kc, int lane) {
  int kb = (lane & 16) ? 8 : 0;
  const __bf16* p = base + (size_t)row * rs + kc * 32 + kb;
  v8bf lo = *(const v8bf*)p;         // k = kc*32 + kb + 0..7
  v8bf hi = *(const v8bf*)(p + 16);  // k = kc*32 + kb + 16..23
  return __builtin_shufflevector(lo, hi, 0, 1, 2, 3, 4, 5, 6, 7,
                                 8, 9, 10, 11, 12, 13, 14, 15);
}

// B-matrix fragment: packed tiles, 32B contiguous per lane (global or LDS).
static __device__ __forceinline__ v16bf load_b_frag(const __bf16* tiled,
                                                    int kc, int lane) {
  const __bf16* p = tiled + (size_t)kc * 512 + lane * 16;
  return *(const v16bf*)p;
}

#ifdef HAVE_TDM
// TDM copy of `nelem` contiguous bf16 elements (nelem < 65536) from global to
// LDS.  D# per CDNA5 ISA ch.8: group0 = {count=1, lds_addr, global_addr,
// type=2}; group1 = {data_size=2B, 1 x nelem tile}; groups 2/3 zero (<=2D).
static __device__ __forceinline__ void tdm_load_tile(unsigned int lds_off,
                                                     const __bf16* gsrc,
                                                     unsigned int nelem) {
  unsigned long long ga = (unsigned long long)(__UINTPTR_TYPE__)gsrc;
  u32x4 g0;
  g0[0] = 1u;                                            // count = 1
  g0[1] = lds_off;                                       // lds_addr
  g0[2] = (unsigned int)(ga & 0xffffffffull);            // global_addr[31:0]
  g0[3] = (unsigned int)((ga >> 32) & 0x1ffffffull) |    // global_addr[56:32]
          (2u << 30);                                    // type = 2 ("image")
  i32x8 g1;
  g1[0] = 1 << 16;              // workgroup_mask=0, data_size=1 (2 bytes)
  g1[1] = (int)(nelem << 16);   // tensor_dim0[15:0] (high bits in g1[2])
  g1[2] = 1 << 16;              // tensor_dim0[31:16]=0, tensor_dim1=1
  g1[3] = (int)(nelem << 16);   // tensor_dim1 hi=0, tile_dim0 = nelem
  g1[4] = 1;                    // tile_dim1 = 1, tile_dim2 = 0
  g1[5] = (int)nelem;           // tensor_dim0_stride[31:0]
  g1[6] = 0;                    // stride hi, tensor_dim1_stride lo
  g1[7] = 0;
  i32x4 z4 = {};
#if defined(__clang_major__) && __clang_major__ >= 23
  i32x8 z8 = {};
  __builtin_amdgcn_tensor_load_to_lds(g0, g1, z4, z4, z8, 0);
#else
  __builtin_amdgcn_tensor_load_to_lds(g0, g1, z4, z4, 0);
#endif
}
#endif

// Stage nelem bf16 from global into LDS: TDM (wave 0) when available,
// cooperative copy otherwise.  Ends with a workgroup barrier.
static __device__ __forceinline__ void stage_tile(__bf16* lds,
                                                  const __bf16* gsrc,
                                                  unsigned int nelem,
                                                  int nthreads) {
#ifdef HAVE_TDM
  if (threadIdx.x < 32) {
    tdm_load_tile((unsigned int)(__UINTPTR_TYPE__)lds, gsrc, nelem);
    __builtin_amdgcn_s_wait_tensorcnt(0);
  }
  __syncthreads();
#else
  for (unsigned int i = threadIdx.x; i < nelem / 8; i += nthreads)
    ((v8bf*)lds)[i] = ((const v8bf*)gsrc)[i];
  __syncthreads();
#endif
}

// ---------------------------------------------------------------------------
// Pack fp32 (K x N) [or transposed (N x K)] matrix into bf16 WMMA B tiles.
// dst element index = ((nt*KC + kc)*32 + lane)*16 + e, with
//   k = kc*32 + (e/8)*16 + ((lane&16)?8:0) + (e%8), n = nt*16 + (lane%16).
// ---------------------------------------------------------------------------
__global__ void pack_bf16_tiles(const float* __restrict__ src,
                                __bf16* __restrict__ dst, int K, int N,
                                int transpose, int total) {
  int KC = K >> 5;
  for (int idx = blockIdx.x * blockDim.x + threadIdx.x; idx < total;
       idx += gridDim.x * blockDim.x) {
    int e = idx & 15;
    int lane = (idx >> 4) & 31;
    int rest = idx >> 9;
    int kc = rest % KC;
    int nt = rest / KC;
    int kb = (lane & 16) ? 8 : 0;
    int k = kc * 32 + (e >> 3) * 16 + kb + (e & 7);
    int n = nt * 16 + (lane & 15);
    float v = transpose ? src[(size_t)n * K + k] : src[(size_t)k * N + n];
    dst[idx] = (__bf16)v;
  }
}

// x_bf[(t*8+b)][e] = bf16(embedding[inputs[b][t]][e]);  2048 x 128 row-major.
__global__ void embed_gather(const int* __restrict__ inputs,
                             const float* __restrict__ emb,
                             __bf16* __restrict__ xbf) {
  int idx = blockIdx.x * blockDim.x + threadIdx.x;
  if (idx >= 2048 * 128) return;
  int e = idx & 127;
  int row = idx >> 7;  // t*8 + b
  int t = row >> 3, b = row & 7;
  int tok = inputs[b * 256 + t];
  xbf[idx] = (__bf16)emb[(size_t)tok * 128 + e];
}

// xg[z][t][b][n] = (x @ Wx[z])[t,b,n] + bias[z][n],  z = dir*3 + gate.
__global__ void __launch_bounds__(256)
proj_kernel(const __bf16* __restrict__ xbf, const __bf16* __restrict__ wxT,
            float* __restrict__ xg, const float* b0, const float* b1,
            const float* b2, const float* b3, const float* b4,
            const float* b5) {
  const int KC = 4;  // K = 128
  __shared__ __align__(32) __bf16 btile[KC * 512];  // 4 KB, shared by 8 waves
  int lane = threadIdx.x & 31;
  int wave = threadIdx.x >> 5;
  int ntile = blockIdx.x;               // 0..15
  int mtile = blockIdx.y * 8 + wave;    // 0..127
  int z = blockIdx.z;                   // 0..5
  const __bf16* wt = wxT + (size_t)z * (16 * KC * 512);
  const float* bias = (z == 0) ? b0 : (z == 1) ? b1 : (z == 2) ? b2
                     : (z == 3) ? b3 : (z == 4) ? b4 : b5;
  stage_tile(btile, wt + (size_t)ntile * KC * 512, KC * 512, 256);

  int arow = mtile * 16 + (lane & 15);
  v8f acc = {};
  for (int kc = 0; kc < KC; ++kc) {
    v16bf a = load_a_frag(xbf, 128, arow, kc, lane);
    v16bf b = load_b_frag(btile, kc, lane);
    acc = wmma_bf16(a, b, acc);
  }
  int ncol = ntile * 16 + (lane & 15);
  float bv = bias[ncol];
  int mbase = (lane & 16) ? 8 : 0;
  for (int v = 0; v < 8; ++v) {
    int row = mtile * 16 + mbase + v;  // = t*8 + b
    int t = row >> 3, bb = row & 7;
    xg[(((size_t)z * 256 + t) * 8 + bb) * 256 + ncol] = acc[v] + bv;
  }
}

// Sequential bidirectional GRU scan. One block per direction, 16 waves.
__global__ void __launch_bounds__(512)
scan_kernel(const __bf16* __restrict__ whT, const float* __restrict__ xg,
            const float* __restrict__ hp_f, const float* __restrict__ hp_b,
            __bf16* __restrict__ hcat) {
  const int KC = 8;   // K = 256
  const int RS = 264; // row stride (elements), multiple of 8
  __shared__ __align__(16) float  h_f[16 * RS];   // h (f32), rows 8..15 = 0 pad
  __shared__ __align__(16) __bf16 h_b[16 * RS];   // h (bf16) = WMMA A source
  __shared__ __align__(16) __bf16 rh_b[16 * RS];  // r*h (bf16) = A for Whh
  __shared__ __align__(16) float  z_f[8 * RS];    // z gate
  int dir = blockIdx.x;
  int tid = threadIdx.x;
  int lane = tid & 31, wave = tid >> 5;
  const float* hp = dir ? hp_b : hp_f;

  for (int i = tid; i < 16 * RS; i += 512) {
    h_f[i] = 0.f; h_b[i] = (__bf16)0.f; rh_b[i] = (__bf16)0.f;
  }
  for (int i = tid; i < 8 * RS; i += 512) z_f[i] = 0.f;
  __syncthreads();
  for (int i = tid; i < 2048; i += 512) {
    int m = i >> 8, n = i & 255;
    float v = hp[i];
    h_f[m * RS + n] = v;
    h_b[m * RS + n] = (__bf16)v;
  }
  __syncthreads();

  int ntile = wave;
  int ncol = ntile * 16 + (lane & 15);
  int arow = lane & 15;

  for (int s = 0; s < 256; ++s) {
    int tt = dir ? (255 - s) : s;  // xg index AND output position

    // Prefetch next step's xg slices (3 x 8 KB) behind this step's WMMA work.
    if (s + 1 < 256) {
      int tn = dir ? (254 - s) : (s + 1);
      for (int g = 0; g < 3; ++g) {
        const float* nx = xg + ((size_t)(dir * 3 + g) * 256 + tn) * 2048;
        __builtin_prefetch(nx + tid * 4, 0, 1);  // 512 thr x 16B = 8 KB
      }
    }

    // Phase A: r and z gates (h @ Whr, h @ Whz) + sigmoid.
    for (int g = 0; g < 2; ++g) {
      const __bf16* wt = whT + (size_t)(dir * 3 + g) * (16 * KC * 512)
                       + (size_t)ntile * KC * 512;
      v8f acc = {};
      for (int kc = 0; kc < KC; ++kc) {
        v16bf a = load_a_frag(h_b, RS, arow, kc, lane);
        v16bf b = load_b_frag(wt, kc, lane);
        acc = wmma_bf16(a, b, acc);
      }
      const float* xgp = xg + ((size_t)(dir * 3 + g) * 256 + tt) * 8 * 256;
      if (lane < 16) {  // lanes 0..15 hold batch rows m = 0..7
        for (int v = 0; v < 8; ++v) {
          float pre = acc[v] + xgp[v * 256 + ncol];
          float sg = 1.f / (1.f + __expf(-pre));
          if (g == 0) {
            rh_b[v * RS + ncol] = (__bf16)(sg * h_f[v * RS + ncol]);
          } else {
            z_f[v * RS + ncol] = sg;
          }
        }
      }
    }
    __syncthreads();

    // Phase B: h~ = tanh((r*h) @ Whh + xh); h = z*h~ + (1-z)*h.
    {
      const __bf16* wt = whT + (size_t)(dir * 3 + 2) * (16 * KC * 512)
                       + (size_t)ntile * KC * 512;
      v8f acc = {};
      for (int kc = 0; kc < KC; ++kc) {
        v16bf a = load_a_frag(rh_b, RS, arow, kc, lane);
        v16bf b = load_b_frag(wt, kc, lane);
        acc = wmma_bf16(a, b, acc);
      }
      const float* xgp = xg + ((size_t)(dir * 3 + 2) * 256 + tt) * 8 * 256;
      if (lane < 16) {
        for (int v = 0; v < 8; ++v) {
          float pre = acc[v] + xgp[v * 256 + ncol];
          float ht = tanhf(pre);
          float zz = z_f[v * RS + ncol];
          float hn = zz * ht + (1.f - zz) * h_f[v * RS + ncol];
          h_f[v * RS + ncol] = hn;
          h_b[v * RS + ncol] = (__bf16)hn;
          // hcat row = b*256 + t, col = dir*256 + n  (bf16, FC A-matrix)
          hcat[((size_t)v * 256 + tt) * 512 + dir * 256 + ncol] = (__bf16)hn;
        }
      }
    }
    __syncthreads();
  }
}

// out[(b*256+t), v] = hcat @ fc_w^T + fc_b;  2048 x 32000 f32.
__global__ void __launch_bounds__(256)
fc_kernel(const __bf16* __restrict__ hcat, const __bf16* __restrict__ fwT,
          const float* __restrict__ fcb, float* __restrict__ out) {
  const int KC = 16;  // K = 512
  __shared__ __align__(32) __bf16 btile[KC * 512];  // 16 KB, shared by 8 waves
  int lane = threadIdx.x & 31, wave = threadIdx.x >> 5;
  int ntile = blockIdx.x;             // 0..1999
  int mtile = blockIdx.y * 8 + wave;  // 0..127
  stage_tile(btile, fwT + (size_t)ntile * KC * 512, KC * 512, 256);

  int arow = mtile * 16 + (lane & 15);
  v8f acc = {};
  for (int kc = 0; kc < KC; ++kc) {
    v16bf a = load_a_frag(hcat, 512, arow, kc, lane);
    v16bf b = load_b_frag(btile, kc, lane);
    acc = wmma_bf16(a, b, acc);
  }
  int n = ntile * 16 + (lane & 15);
  float bv = fcb[n];
  int mbase = mtile * 16 + ((lane & 16) ? 8 : 0);
  for (int v = 0; v < 8; ++v) {
    out[(size_t)(mbase + v) * 32000 + n] = acc[v] + bv;
  }
}

// ---------------------------------------------------------------------------
extern "C" void kernel_launch(void* const* d_in, const int* in_sizes, int n_in,
                              void* d_out, int out_size, void* d_ws,
                              size_t ws_size, hipStream_t stream) {
  const int*   inputs = (const int*)d_in[0];
  const float* hp_f   = (const float*)d_in[1];
  const float* hp_b   = (const float*)d_in[2];
  const float* emb    = (const float*)d_in[3];
  // z = dir*3 + gate (gate: 0=r,1=z,2=h)
  const float* Wh[6] = {(const float*)d_in[4],  (const float*)d_in[7],
                        (const float*)d_in[10], (const float*)d_in[13],
                        (const float*)d_in[16], (const float*)d_in[19]};
  const float* Wx[6] = {(const float*)d_in[5],  (const float*)d_in[8],
                        (const float*)d_in[11], (const float*)d_in[14],
                        (const float*)d_in[17], (const float*)d_in[20]};
  const float* bb[6] = {(const float*)d_in[6],  (const float*)d_in[9],
                        (const float*)d_in[12], (const float*)d_in[15],
                        (const float*)d_in[18], (const float*)d_in[21]};
  const float* fcw = (const float*)d_in[22];
  const float* fcb = (const float*)d_in[23];
  float* out = (float*)d_out;

  char* ws = (char*)d_ws;
  __bf16* xbf  = (__bf16*)(ws + 0);          // 2048*128 bf16        (512 KB)
  __bf16* wxT  = (__bf16*)(ws + 524288);     // 6 * 32768 bf16       (384 KB)
  __bf16* whT  = (__bf16*)(ws + 917504);     // 6 * 65536 bf16       (768 KB)
  __bf16* fwT  = (__bf16*)(ws + 1703936);    // 2000*16*512 bf16     (~31 MB)
  float*  xg   = (float*)(ws + 34471936);    // 6*256*8*256 f32      (12 MB)
  __bf16* hcat = (__bf16*)(ws + 47054848);   // 2048*512 bf16        (2 MB)

  // 1) Pack weights into WMMA B-tile layout (bf16).
  for (int z = 0; z < 6; ++z) {
    pack_bf16_tiles<<<128, 256, 0, stream>>>(Wx[z], wxT + (size_t)z * 32768,
                                             128, 256, 0, 32768);
    pack_bf16_tiles<<<256, 256, 0, stream>>>(Wh[z], whT + (size_t)z * 65536,
                                             256, 256, 0, 65536);
  }
  pack_bf16_tiles<<<8192, 256, 0, stream>>>(fcw, fwT, 512, 32000, 1, 16384000);

  // 2) Embedding gather -> bf16 A-matrix for projections.
  embed_gather<<<1024, 256, 0, stream>>>(inputs, emb, xbf);

  // 3) Input projections xg = x @ Wx + b (all 6 gate/dir combos).
  proj_kernel<<<dim3(16, 16, 6), 256, 0, stream>>>(
      xbf, wxT, xg, bb[0], bb[1], bb[2], bb[3], bb[4], bb[5]);

  // 4) Bidirectional GRU scan (fwd = block 0, bwd = block 1).
  scan_kernel<<<2, 512, 0, stream>>>(whT, xg, hp_f, hp_b, hcat);

  // 5) Final FC: out = hcat @ fc_w^T + fc_b.
  fc_kernel<<<dim3(2000, 16), 256, 0, stream>>>(hcat, fwT, fcb, out);
}
